// Net_26860725469743
// MI455X (gfx1250) — matrix operand order repaired and checked by reading, not compiled, not run on hardware
//
#include <hip/hip_runtime.h>
#include <hip/hip_bf16.h>

// ---------------------------------------------------------------------------
// DGCNN-style net on gfx1250 (MI455X): knn -> EdgeConv(6->64^3) -> knn ->
// EdgeConv(128->128,128,256) -> lin0+maxpool -> fused lin1+lin2.
// GEMMs run on v_wmma_f32_16x16x32_bf16; weight tiles are staged into LDS with
// GLOBAL_LOAD_ASYNC_TO_LDS_B128 (ASYNCcnt) from pre-transposed bf16 copies.
// ---------------------------------------------------------------------------

#define DEVINL __device__ __forceinline__

typedef __attribute__((ext_vector_type(16))) __bf16 v16bf;
typedef __attribute__((ext_vector_type(8)))  __bf16 v8bf;
typedef __attribute__((ext_vector_type(8)))  float  v8f;

union Frag { v16bf v; v8bf h[2]; unsigned u[8]; };

static constexpr int NB   = 8;
static constexpr int P    = 2048;
static constexpr int NPTS = NB * P;   // 16384
static constexpr int KNN  = 30;

// ---------------------------------------------------------------------------
// WMMA helpers.
//
// A-fragment (16x32 bf16, MxK): lane L, m = L&15 (row), h = L>>4.
//   element e<8  -> K = kbase + 8h + e        (contiguous 8 bf16 = 16B)
//   element e>=8 -> K = kbase + 16 + 8h + (e-8)
// B-fragment (32x16 bf16, KxN) from W stored *transposed* [N][ldk]:
//   lane L, n = L&15 (col), h = L>>4, same K order as the A-fragment
// C/D (16x16 f32): vgpr r, lane L -> row M = r + 8h, col N = L&15.
// ---------------------------------------------------------------------------

DEVINL v8f wmma_bf16(v16bf a, v16bf b, v8f c) {
  return __builtin_amdgcn_wmma_f32_16x16x32_bf16(false, a, false, b, (short)0, c,
                                                 false, false);
}

DEVINL unsigned pack_bf16(float lo, float hi) {
  union { unsigned u; __bf16 b[2]; } p;
  p.b[0] = (__bf16)lo; p.b[1] = (__bf16)hi;
  return p.u;
}

DEVINL v16bf load_a_lds(const __bf16* Xl, int ldk, int kbase) {
  int lane = threadIdx.x & 31;
  int m = lane & 15, hh = lane >> 4;
  const __bf16* p = Xl + m * ldk + kbase + 8 * hh;
  Frag f;
  f.h[0] = *(const v8bf*)(p);
  f.h[1] = *(const v8bf*)(p + 16);
  return f.v;
}

DEVINL v16bf load_b_lds(const __bf16* Wt, int ldk, int nbase, int kbase) {
  int lane = threadIdx.x & 31;
  int n = lane & 15, hh = lane >> 4;
  const __bf16* p = Wt + (nbase + n) * ldk + kbase + 8 * hh;
  Frag f;
  f.h[0] = *(const v8bf*)(p);
  f.h[1] = *(const v8bf*)(p + 16);
  return f.v;
}

// A fragment from a global f32 row-major matrix (float4 loads + pack).
DEVINL v16bf load_a_gf32(const float* __restrict__ X, int ldk, int rowbase, int kbase) {
  int lane = threadIdx.x & 31;
  int m = lane & 15, hh = lane >> 4;
  const float4* p = (const float4*)(X + (size_t)(rowbase + m) * ldk + kbase + 8 * hh);
  float4 v0 = p[0], v1 = p[1], v2 = p[4], v3 = p[5];
  Frag f;
  f.u[0] = pack_bf16(v0.x, v0.y); f.u[1] = pack_bf16(v0.z, v0.w);
  f.u[2] = pack_bf16(v1.x, v1.y); f.u[3] = pack_bf16(v1.z, v1.w);
  f.u[4] = pack_bf16(v2.x, v2.y); f.u[5] = pack_bf16(v2.z, v2.w);
  f.u[6] = pack_bf16(v3.x, v3.y); f.u[7] = pack_bf16(v3.z, v3.w);
  return f.v;
}

DEVINL v8f splat_bias(const float* __restrict__ b, int nbase) {
  float bv = b[nbase + (threadIdx.x & 15)];
  v8f acc;
#pragma unroll
  for (int r = 0; r < 8; ++r) acc[r] = bv;
  return acc;
}

DEVINL void relu8(v8f& a) {
#pragma unroll
  for (int r = 0; r < 8; ++r) a[r] = fmaxf(a[r], 0.0f);
}

DEVINL void store_c_lds_bf16(__bf16* Hl, int ldh, int nbase, v8f acc) {
  int lane = threadIdx.x & 31;
  int n = lane & 15, hh = lane >> 4;
#pragma unroll
  for (int r = 0; r < 8; ++r)
    Hl[(r + 8 * hh) * ldh + nbase + n] = (__bf16)acc[r];
}

// Column-max over the 16 rows of a C tile; valid in all lanes for col (lane&15).
DEVINL float cmax_reduce(v8f acc) {
  float m = acc[0];
#pragma unroll
  for (int r = 1; r < 8; ++r) m = fmaxf(m, acc[r]);
  m = fmaxf(m, __shfl_xor(m, 16, 32));
  return m;
}

// ---------------------------------------------------------------------------
// Async global->LDS staging (CDNA5 GLOBAL_LOAD_ASYNC_TO_LDS_B128, ASYNCcnt).
// Flat pointer to __shared__ carries the LDS byte offset in its low 32 bits.
// ---------------------------------------------------------------------------

DEVINL void async_copy_b128(const __bf16* gsrc, __bf16* ldst) {
  unsigned lds = (unsigned)(uintptr_t)(void*)ldst;
  unsigned long long ga = (unsigned long long)(uintptr_t)gsrc;
  asm volatile("global_load_async_to_lds_b128 %0, %1, off"
               :: "v"(lds), "v"(ga) : "memory");
}

DEVINL void wait_async() {
  asm volatile("s_wait_asynccnt 0" ::: "memory");
}

// Copy n_b128 16B packets from a contiguous bf16 global region into LDS.
DEVINL void stage_async(const __bf16* __restrict__ gsrc, __bf16* ldst, int n_b128,
                        int tid, int nthreads) {
  for (int i = tid; i < n_b128; i += nthreads)
    async_copy_b128(gsrc + (size_t)i * 8, ldst + (size_t)i * 8);
}

// ---------------------------------------------------------------------------
// Weight prep: f32 [Kd][Nd] -> transposed bf16 [Nd][ldk], zero-padded K.
// ---------------------------------------------------------------------------

__global__ __launch_bounds__(256) void prep_wt_kernel(const float* __restrict__ W,
                                                      __bf16* __restrict__ Wt,
                                                      int Kd, int Nd, int ldk) {
  int idx = blockIdx.x * 256 + threadIdx.x;
  if (idx >= Nd * ldk) return;
  int n = idx / ldk, k = idx % ldk;
  Wt[idx] = (k < Kd) ? (__bf16)W[(size_t)k * Nd + n] : (__bf16)0.0f;
}

// ---------------------------------------------------------------------------
// kNN kernels: thread-per-point insertion-sorted top-30, LDS candidate tiles,
// next tile prefetched (global_prefetch_b8) while current one is processed.
// ---------------------------------------------------------------------------

__global__ __launch_bounds__(256) void knn3_kernel(const float* __restrict__ pos,
                                                   int* __restrict__ nbr) {
  int gid  = blockIdx.x * 256 + threadIdx.x;
  int base = gid & ~(P - 1);
  float qx = pos[(size_t)gid * 3 + 0];
  float qy = pos[(size_t)gid * 3 + 1];
  float qz = pos[(size_t)gid * 3 + 2];

  float bd[KNN];
  int   bi[KNN];
#pragma unroll
  for (int t = 0; t < KNN; ++t) { bd[t] = 3.0e38f; bi[t] = base; }

  __shared__ float tx[256], ty[256], tz[256];
  for (int t0 = 0; t0 < P; t0 += 256) {
    __syncthreads();
    int c = base + t0 + threadIdx.x;
    tx[threadIdx.x] = pos[(size_t)c * 3 + 0];
    ty[threadIdx.x] = pos[(size_t)c * 3 + 1];
    tz[threadIdx.x] = pos[(size_t)c * 3 + 2];
    __syncthreads();
    if (t0 + 256 < P)
      __builtin_prefetch(pos + (size_t)(base + t0 + 256 + threadIdx.x) * 3, 0, 0);
    for (int u = 0; u < 256; ++u) {
      int cand = base + t0 + u;
      float dx = tx[u] - qx, dy = ty[u] - qy, dz = tz[u] - qz;
      float d = dx * dx + dy * dy + dz * dz;
      if (cand != gid && d < bd[KNN - 1]) {
        int t = KNN - 1;
        while (t > 0 && bd[t - 1] > d) { bd[t] = bd[t - 1]; bi[t] = bi[t - 1]; --t; }
        bd[t] = d; bi[t] = cand;
      }
    }
  }
  for (int t = 0; t < KNN; ++t) nbr[(size_t)gid * KNN + t] = bi[t];
}

__global__ __launch_bounds__(256) void knn64_kernel(const float* __restrict__ x,
                                                    int* __restrict__ nbr) {
  int gid  = blockIdx.x * 256 + threadIdx.x;
  int base = gid & ~(P - 1);

  float4 q[16];
#pragma unroll
  for (int t = 0; t < 16; ++t) q[t] = ((const float4*)(x + (size_t)gid * 64))[t];

  float bd[KNN];
  int   bi[KNN];
#pragma unroll
  for (int t = 0; t < KNN; ++t) { bd[t] = 3.0e38f; bi[t] = base; }

  __shared__ float4 tile[64 * 16];  // 64 candidates x 64 feats = 16KB
  for (int t0 = 0; t0 < P; t0 += 64) {
    __syncthreads();
    for (int idx = threadIdx.x; idx < 64 * 16; idx += 256) {
      int r = idx >> 4, cc = idx & 15;
      tile[idx] = ((const float4*)(x + (size_t)(base + t0 + r) * 64))[cc];
    }
    __syncthreads();
    if (t0 + 64 < P) {
      int r = threadIdx.x & 63, seg = threadIdx.x >> 6;
      __builtin_prefetch(x + (size_t)(base + t0 + 64 + r) * 64 + seg * 16, 0, 0);
    }
    for (int u = 0; u < 64; ++u) {
      float d = 0.0f;
#pragma unroll
      for (int cc = 0; cc < 16; ++cc) {
        float4 tv = tile[u * 16 + cc];
        float dx = tv.x - q[cc].x; d += dx * dx;
        float dy = tv.y - q[cc].y; d += dy * dy;
        float dz = tv.z - q[cc].z; d += dz * dz;
        float dw = tv.w - q[cc].w; d += dw * dw;
      }
      int cand = base + t0 + u;
      if (cand != gid && d < bd[KNN - 1]) {
        int t = KNN - 1;
        while (t > 0 && bd[t - 1] > d) { bd[t] = bd[t - 1]; bi[t] = bi[t - 1]; --t; }
        bd[t] = d; bi[t] = cand;
      }
    }
  }
  for (int t = 0; t < KNN; ++t) nbr[(size_t)gid * KNN + t] = bi[t];
}

// ---------------------------------------------------------------------------
// EdgeConv1: per-point fused MLP 6->64->64->64 + max over 30 edges (pad to 32).
// 4 waves per block, one point per wave; weights async-staged once (~20KB LDS).
// ---------------------------------------------------------------------------

__global__ __launch_bounds__(128) void edgeconv1_kernel(
    const float* __restrict__ pos, const int* __restrict__ nbr,
    const __bf16* __restrict__ W0t_g, const float* __restrict__ b0,
    const __bf16* __restrict__ W1t_g, const float* __restrict__ b1,
    const __bf16* __restrict__ W2t_g, const float* __restrict__ b2,
    float* __restrict__ x1) {
  __shared__ __align__(16) __bf16 W0t[64 * 32];   // [N=64][K=32] (K padded from 6)
  __shared__ __align__(16) __bf16 W1t[64 * 64];
  __shared__ __align__(16) __bf16 W2t[64 * 64];
  __shared__ __align__(16) __bf16 Xs[4][32 * 32]; // per-wave 32 edge rows x 32 K
  __shared__ __align__(16) __bf16 Hs[4][16 * 64]; // per-wave hidden tile

  int tid = threadIdx.x, wave = tid >> 5, lane = tid & 31;
  int point = blockIdx.x * 4 + wave;

  // Kick off async weight staging, overlap with edge gather below.
  stage_async(W0t_g, W0t, 256, tid, 128);
  stage_async(W1t_g, W1t, 512, tid, 128);
  stage_async(W2t_g, W2t, 512, tid, 128);

  // stage X: 32 rows (edges; rows 30/31 replicate edge 29 so max is unchanged)
  {
    int ke = (lane < KNN) ? lane : (KNN - 1);
    int j = nbr[(size_t)point * KNN + ke];
    float xi0 = pos[(size_t)point * 3 + 0];
    float xi1 = pos[(size_t)point * 3 + 1];
    float xi2 = pos[(size_t)point * 3 + 2];
    float xj0 = pos[(size_t)j * 3 + 0];
    float xj1 = pos[(size_t)j * 3 + 1];
    float xj2 = pos[(size_t)j * 3 + 2];
    __bf16* row = &Xs[wave][lane * 32];
#pragma unroll
    for (int c = 6; c < 32; ++c) row[c] = (__bf16)0.0f;
    row[0] = (__bf16)xi0; row[1] = (__bf16)xi1; row[2] = (__bf16)xi2;
    row[3] = (__bf16)(xj0 - xi0); row[4] = (__bf16)(xj1 - xi1); row[5] = (__bf16)(xj2 - xi2);
  }
  wait_async();
  __syncthreads();

  float colmax[4];
#pragma unroll
  for (int nt = 0; nt < 4; ++nt) colmax[nt] = -3.0e38f;

  for (int mt = 0; mt < 2; ++mt) {
    const __bf16* Xl = &Xs[wave][mt * 16 * 32];
    __bf16* Hl = &Hs[wave][0];

    // layer 1: K=32 (1 kt), N=64 (4 nt)
    v16bf a0 = load_a_lds(Xl, 32, 0);
#pragma unroll
    for (int nt = 0; nt < 4; ++nt) {
      v8f acc = splat_bias(b0, nt * 16);
      acc = wmma_bf16(a0, load_b_lds(W0t, 32, nt * 16, 0), acc);
      relu8(acc);
      store_c_lds_bf16(Hl, 64, nt * 16, acc);
    }
    // layer 2: K=64 (2 kt), N=64 — A frags hold all of H, safe to overwrite
    v16bf a1_0 = load_a_lds(Hl, 64, 0);
    v16bf a1_1 = load_a_lds(Hl, 64, 32);
#pragma unroll
    for (int nt = 0; nt < 4; ++nt) {
      v8f acc = splat_bias(b1, nt * 16);
      acc = wmma_bf16(a1_0, load_b_lds(W1t, 64, nt * 16, 0), acc);
      acc = wmma_bf16(a1_1, load_b_lds(W1t, 64, nt * 16, 32), acc);
      relu8(acc);
      store_c_lds_bf16(Hl, 64, nt * 16, acc);
    }
    // layer 3 + max over edges
    v16bf a2_0 = load_a_lds(Hl, 64, 0);
    v16bf a2_1 = load_a_lds(Hl, 64, 32);
#pragma unroll
    for (int nt = 0; nt < 4; ++nt) {
      v8f acc = splat_bias(b2, nt * 16);
      acc = wmma_bf16(a2_0, load_b_lds(W2t, 64, nt * 16, 0), acc);
      acc = wmma_bf16(a2_1, load_b_lds(W2t, 64, nt * 16, 32), acc);
      relu8(acc);
      colmax[nt] = fmaxf(colmax[nt], cmax_reduce(acc));
    }
  }
  if (lane < 16) {
#pragma unroll
    for (int nt = 0; nt < 4; ++nt)
      x1[(size_t)point * 64 + nt * 16 + lane] = colmax[nt];
  }
}

// ---------------------------------------------------------------------------
// EdgeConv2: per-point fused MLP 128->128->128->256 + max over 30 edges.
// Weight chunks ([64 N][128 K] = 16KB) async-staged; LDS ~48KB total.
// ---------------------------------------------------------------------------

__global__ __launch_bounds__(128) void edgeconv2_kernel(
    const float* __restrict__ x1f, const int* __restrict__ nbr,
    const __bf16* __restrict__ W0t_g, const float* __restrict__ b0,
    const __bf16* __restrict__ W1t_g, const float* __restrict__ b1,
    const __bf16* __restrict__ W2t_g, const float* __restrict__ b2,
    float* __restrict__ x2) {
  __shared__ __align__(16) __bf16 WB[64 * 128];    // 16KB weight chunk [N=64][K=128]
  __shared__ __align__(16) __bf16 Xs[4][16 * 128]; // per-wave A tile
  __shared__ __align__(16) __bf16 Hs[4][16 * 128]; // per-wave hidden tile

  int tid = threadIdx.x, wave = tid >> 5, lane = tid & 31;
  int point = blockIdx.x * 4 + wave;
  const int* nrow = nbr + (size_t)point * KNN;

  float colmax[16];
#pragma unroll
  for (int g = 0; g < 16; ++g) colmax[g] = -3.0e38f;

  for (int mt = 0; mt < 2; ++mt) {
    // stage X tile: 16 rows of [xi | xj - xi]; two lanes per row, packed bf16
    {
      int r = lane >> 1, half = lane & 1;
      int k = mt * 16 + r; if (k > KNN - 1) k = KNN - 1;
      int j = nrow[k];
      const float4* pa = (const float4*)(x1f + (size_t)(half ? j : point) * 64);
      const float4* pb = (const float4*)(x1f + (size_t)point * 64);
      uint2* d2 = (uint2*)&Xs[wave][r * 128 + half * 64];
#pragma unroll
      for (int c = 0; c < 16; ++c) {
        float4 a = pa[c];
        if (half) {
          float4 b = pb[c];
          a.x -= b.x; a.y -= b.y; a.z -= b.z; a.w -= b.w;
        }
        uint2 o;
        o.x = pack_bf16(a.x, a.y);
        o.y = pack_bf16(a.z, a.w);
        d2[c] = o;
      }
    }
    __syncthreads();

    v16bf a[4];
    // LAYER 1: 128->128
#pragma unroll
    for (int kt = 0; kt < 4; ++kt) a[kt] = load_a_lds(&Xs[wave][0], 128, kt * 32);
    for (int nc = 0; nc < 2; ++nc) {
      __syncthreads();
      stage_async(W0t_g + (size_t)(nc * 64) * 128, WB, 1024, tid, 128);
      wait_async();
      __syncthreads();
#pragma unroll
      for (int nt = 0; nt < 4; ++nt) {
        v8f acc = splat_bias(b0 + nc * 64, nt * 16);
#pragma unroll
        for (int kt = 0; kt < 4; ++kt)
          acc = wmma_bf16(a[kt], load_b_lds(WB, 128, nt * 16, kt * 32), acc);
        relu8(acc);
        store_c_lds_bf16(&Hs[wave][0], 128, nc * 64 + nt * 16, acc);
      }
    }
    // LAYER 2: 128->128 (A frags hold all of H -> safe to overwrite)
#pragma unroll
    for (int kt = 0; kt < 4; ++kt) a[kt] = load_a_lds(&Hs[wave][0], 128, kt * 32);
    for (int nc = 0; nc < 2; ++nc) {
      __syncthreads();
      stage_async(W1t_g + (size_t)(nc * 64) * 128, WB, 1024, tid, 128);
      wait_async();
      __syncthreads();
#pragma unroll
      for (int nt = 0; nt < 4; ++nt) {
        v8f acc = splat_bias(b1 + nc * 64, nt * 16);
#pragma unroll
        for (int kt = 0; kt < 4; ++kt)
          acc = wmma_bf16(a[kt], load_b_lds(WB, 128, nt * 16, kt * 32), acc);
        relu8(acc);
        store_c_lds_bf16(&Hs[wave][0], 128, nc * 64 + nt * 16, acc);
      }
    }
    // LAYER 3: 128->256 + max over edges
#pragma unroll
    for (int kt = 0; kt < 4; ++kt) a[kt] = load_a_lds(&Hs[wave][0], 128, kt * 32);
    for (int nc = 0; nc < 4; ++nc) {
      __syncthreads();
      stage_async(W2t_g + (size_t)(nc * 64) * 128, WB, 1024, tid, 128);
      wait_async();
      __syncthreads();
#pragma unroll
      for (int nt = 0; nt < 4; ++nt) {
        v8f acc = splat_bias(b2 + nc * 64, nt * 16);
#pragma unroll
        for (int kt = 0; kt < 4; ++kt)
          acc = wmma_bf16(a[kt], load_b_lds(WB, 128, nt * 16, kt * 32), acc);
        relu8(acc);
        int g = nc * 4 + nt;
        colmax[g] = fmaxf(colmax[g], cmax_reduce(acc));
      }
    }
  }
  if (lane < 16) {
#pragma unroll
    for (int g = 0; g < 16; ++g)
      x2[(size_t)point * 256 + g * 16 + lane] = colmax[g];
  }
}

// ---------------------------------------------------------------------------
// lin0 + global max pool: h = relu(x2 @ W[256,512] + b); pooled[b] = max rows.
// ReLU >= 0 so float max == uint-bit max (pooled pre-zeroed).
// ---------------------------------------------------------------------------

__global__ __launch_bounds__(256) void lin0_pool_kernel(
    const float* __restrict__ x2, const __bf16* __restrict__ Wt_g,
    const float* __restrict__ bias, unsigned* __restrict__ pooled) {
  __shared__ __align__(16) __bf16 Wt[64 * 256];  // [N=64][K=256] chunk, 32KB

  int tid = threadIdx.x, wave = tid >> 5, lane = tid & 31;
  int rowbase = blockIdx.x * 128 + wave * 16;
  int b = rowbase >> 11;

  v16bf a[8];
#pragma unroll
  for (int kt = 0; kt < 8; ++kt) a[kt] = load_a_gf32(x2, 256, rowbase, kt * 32);

  for (int nc = 0; nc < 8; ++nc) {
    __syncthreads();
    stage_async(Wt_g + (size_t)(nc * 64) * 256, Wt, 2048, tid, 256);
    wait_async();
    __syncthreads();
#pragma unroll
    for (int nt = 0; nt < 4; ++nt) {
      v8f acc = splat_bias(bias + nc * 64, nt * 16);
#pragma unroll
      for (int kt = 0; kt < 8; ++kt)
        acc = wmma_bf16(a[kt], load_b_lds(Wt, 256, nt * 16, kt * 32), acc);
      relu8(acc);
      float cm = cmax_reduce(acc);
      if (lane < 16)
        atomicMax(&pooled[(size_t)b * 512 + nc * 64 + nt * 16 + lane],
                  __float_as_uint(cm));
    }
  }
}

// ---------------------------------------------------------------------------
// Head: x = relu(cat[x1|x2|g] @ lin1[832,256]); out = relu(x @ lin2[256,256]).
// Fused; lin1 streamed in K-slices of 32, lin2 in N-chunks of 32.
// ---------------------------------------------------------------------------

DEVINL v16bf load_a_concat(const float* __restrict__ x1f, const float* __restrict__ x2f,
                           const unsigned* __restrict__ pooled, int rowbase, int b,
                           int kbase) {
  int lane = threadIdx.x & 31;
  int m = lane & 15, hh = lane >> 4;
  int row = rowbase + m;
  Frag f;
#pragma unroll
  for (int part = 0; part < 2; ++part) {
    int k0 = kbase + 8 * hh + part * 16;
#pragma unroll
    for (int e = 0; e < 8; ++e) {
      int k = k0 + e;
      float v;
      if (k < 64)       v = x1f[(size_t)row * 64 + k];
      else if (k < 320) v = x2f[(size_t)row * 256 + (k - 64)];
      else              v = __uint_as_float(pooled[(size_t)b * 512 + (k - 320)]);
      f.h[part][e] = (__bf16)v;
    }
  }
  return f.v;
}

__global__ __launch_bounds__(128) void head_kernel(
    const float* __restrict__ x1f, const float* __restrict__ x2f,
    const unsigned* __restrict__ pooled,
    const __bf16* __restrict__ W1t_g, const float* __restrict__ bb1,
    const __bf16* __restrict__ W2t_g, const float* __restrict__ bb2,
    float* __restrict__ out) {
  __shared__ __align__(16) __bf16 WB[256 * 32];    // 16KB slice/chunk
  __shared__ __align__(16) __bf16 Hs[4][16 * 256]; // 32KB hidden

  int tid = threadIdx.x, wave = tid >> 5, lane = tid & 31;
  int rowbase = blockIdx.x * 64 + wave * 16;
  int b = rowbase >> 11;

  v8f acc[16];
#pragma unroll
  for (int nt = 0; nt < 16; ++nt) acc[nt] = splat_bias(bb1, nt * 16);

  for (int ks = 0; ks < 26; ++ks) {  // 832 / 32 K-slices
    __syncthreads();
    // W1t is [256 n][832 k]: slice rows are strided, 4x16B packets per row.
    for (int i = tid; i < 1024; i += 128) {
      int n = i >> 2, q = i & 3;
      async_copy_b128(W1t_g + (size_t)n * 832 + ks * 32 + q * 8, WB + (size_t)i * 8);
    }
    wait_async();
    __syncthreads();
    v16bf a = load_a_concat(x1f, x2f, pooled, rowbase, b, ks * 32);
#pragma unroll
    for (int nt = 0; nt < 16; ++nt)
      acc[nt] = wmma_bf16(a, load_b_lds(WB, 32, nt * 16, 0), acc[nt]);
  }
#pragma unroll
  for (int nt = 0; nt < 16; ++nt) {
    relu8(acc[nt]);
    store_c_lds_bf16(&Hs[wave][0], 256, nt * 16, acc[nt]);
  }

  v16bf ah[8];
#pragma unroll
  for (int kt = 0; kt < 8; ++kt) ah[kt] = load_a_lds(&Hs[wave][0], 256, kt * 32);

  for (int c = 0; c < 8; ++c) {  // 256 cols in chunks of 32
    __syncthreads();
    stage_async(W2t_g + (size_t)(c * 32) * 256, WB, 1024, tid, 128);
    wait_async();
    __syncthreads();
#pragma unroll
    for (int nt = 0; nt < 2; ++nt) {
      v8f a2 = splat_bias(bb2 + c * 32, nt * 16);
#pragma unroll
      for (int kt = 0; kt < 8; ++kt)
        a2 = wmma_bf16(ah[kt], load_b_lds(WB, 256, nt * 16, kt * 32), a2);
      relu8(a2);
      int ncol = c * 32 + nt * 16 + (lane & 15);
      int hh = lane >> 4;
#pragma unroll
      for (int r = 0; r < 8; ++r)
        out[(size_t)(rowbase + r + 8 * hh) * 256 + ncol] = a2[r];
    }
  }
}

// ---------------------------------------------------------------------------
// Launch
// ---------------------------------------------------------------------------

static inline int cdiv(int a, int b) { return (a + b - 1) / b; }

extern "C" void kernel_launch(void* const* d_in, const int* in_sizes, int n_in,
                              void* d_out, int out_size, void* d_ws, size_t ws_size,
                              hipStream_t stream) {
  (void)in_sizes; (void)n_in; (void)out_size; (void)ws_size;

  const float* pos   = (const float*)d_in[0];
  const float* c1W0  = (const float*)d_in[2];
  const float* c1b0  = (const float*)d_in[3];
  const float* c1W1  = (const float*)d_in[4];
  const float* c1b1  = (const float*)d_in[5];
  const float* c1W2  = (const float*)d_in[6];
  const float* c1b2  = (const float*)d_in[7];
  const float* c2W0  = (const float*)d_in[8];
  const float* c2b0  = (const float*)d_in[9];
  const float* c2W1  = (const float*)d_in[10];
  const float* c2b1  = (const float*)d_in[11];
  const float* c2W2  = (const float*)d_in[12];
  const float* c2b2  = (const float*)d_in[13];
  const float* l0W   = (const float*)d_in[14];
  const float* l0b   = (const float*)d_in[15];
  const float* l1W   = (const float*)d_in[16];
  const float* l1b   = (const float*)d_in[17];
  const float* l2W   = (const float*)d_in[18];
  const float* l2b   = (const float*)d_in[19];
  float* out = (float*)d_out;

  char* ws = (char*)d_ws;
  int* nbr1 = (int*)ws;            ws += (size_t)NPTS * KNN * sizeof(int);
  int* nbr2 = (int*)ws;            ws += (size_t)NPTS * KNN * sizeof(int);
  float* x1 = (float*)ws;          ws += (size_t)NPTS * 64 * sizeof(float);
  float* x2 = (float*)ws;          ws += (size_t)NPTS * 256 * sizeof(float);
  unsigned* pooled = (unsigned*)ws; ws += (size_t)NB * 512 * sizeof(unsigned);
  __bf16* c1w0t = (__bf16*)ws;     ws += (size_t)64 * 32 * 2;
  __bf16* c1w1t = (__bf16*)ws;     ws += (size_t)64 * 64 * 2;
  __bf16* c1w2t = (__bf16*)ws;     ws += (size_t)64 * 64 * 2;
  __bf16* c2w0t = (__bf16*)ws;     ws += (size_t)128 * 128 * 2;
  __bf16* c2w1t = (__bf16*)ws;     ws += (size_t)128 * 128 * 2;
  __bf16* c2w2t = (__bf16*)ws;     ws += (size_t)256 * 128 * 2;
  __bf16* l0wt  = (__bf16*)ws;     ws += (size_t)512 * 256 * 2;
  __bf16* l1wt  = (__bf16*)ws;     ws += (size_t)256 * 832 * 2;
  __bf16* l2wt  = (__bf16*)ws;     ws += (size_t)256 * 256 * 2;

  // One-shot weight transpose/convert to bf16.
  prep_wt_kernel<<<cdiv(64 * 32, 256), 256, 0, stream>>>(c1W0, c1w0t, 6, 64, 32);
  prep_wt_kernel<<<cdiv(64 * 64, 256), 256, 0, stream>>>(c1W1, c1w1t, 64, 64, 64);
  prep_wt_kernel<<<cdiv(64 * 64, 256), 256, 0, stream>>>(c1W2, c1w2t, 64, 64, 64);
  prep_wt_kernel<<<cdiv(128 * 128, 256), 256, 0, stream>>>(c2W0, c2w0t, 128, 128, 128);
  prep_wt_kernel<<<cdiv(128 * 128, 256), 256, 0, stream>>>(c2W1, c2w1t, 128, 128, 128);
  prep_wt_kernel<<<cdiv(256 * 128, 256), 256, 0, stream>>>(c2W2, c2w2t, 128, 256, 128);
  prep_wt_kernel<<<cdiv(512 * 256, 256), 256, 0, stream>>>(l0W, l0wt, 256, 512, 256);
  prep_wt_kernel<<<cdiv(256 * 832, 256), 256, 0, stream>>>(l1W, l1wt, 832, 256, 832);
  prep_wt_kernel<<<cdiv(256 * 256, 256), 256, 0, stream>>>(l2W, l2wt, 256, 256, 256);

  knn3_kernel<<<NPTS / 256, 256, 0, stream>>>(pos, nbr1);
  edgeconv1_kernel<<<NPTS / 4, 128, 0, stream>>>(pos, nbr1, c1w0t, c1b0, c1w1t, c1b1,
                                                 c1w2t, c1b2, x1);
  knn64_kernel<<<NPTS / 256, 256, 0, stream>>>(x1, nbr2);
  edgeconv2_kernel<<<NPTS / 4, 128, 0, stream>>>(x1, nbr2, c2w0t, c2b0, c2w1t, c2b1,
                                                 c2w2t, c2b2, x2);
  hipMemsetAsync(pooled, 0, (size_t)NB * 512 * sizeof(unsigned), stream);
  lin0_pool_kernel<<<NPTS / 128, 256, 0, stream>>>(x2, l0wt, l0b, pooled);
  head_kernel<<<NPTS / 64, 128, 0, stream>>>(x1, x2, pooled, l1wt, l1b, l2wt, l2b, out);
}
